// GNNObservationEncoder_31190052504015
// MI455X (gfx1250) — matrix operand order repaired and checked by reading, not compiled, not run on hardware
//
#include <hip/hip_runtime.h>
#include <hip/hip_bf16.h>

// ---------------------------------------------------------------------------
// GAT (2-layer, multi-head) for MI455X / gfx1250.
// B=8, N=1024, OBS=64, HID=192, H=3, D=64.
// GEMMs via v_wmma_f32_16x16x32_f16 (f16 in, f32 accum); softmax in f32 VALU.
// ---------------------------------------------------------------------------

typedef __attribute__((ext_vector_type(16))) _Float16 v16h;
typedef __attribute__((ext_vector_type(8)))  _Float16 v8h;
typedef __attribute__((ext_vector_type(8)))  float    v8f;

#define GAT_B   8
#define GAT_N   1024
#define GAT_H   3
#define GAT_D   64
#define GAT_HID 192
#define NEGV    (-9.0e15f)
#define ALPHA   0.2f

// ---------------------------------------------------------------------------
// f32 -> f16 elementwise convert (layer-1 input features)
// ---------------------------------------------------------------------------
__global__ void cvt_f32_to_f16_kernel(const float* __restrict__ x,
                                      _Float16* __restrict__ y, int n) {
  int i = blockIdx.x * 256 + threadIdx.x;
  if (i < n) y[i] = (_Float16)x[i];
}

// ---------------------------------------------------------------------------
// W [H][Fin][D] f32  ->  Wt [H][D][Fin] f16 (so WMMA B-fragment loads are
// contiguous v16h loads along Fin)
// ---------------------------------------------------------------------------
__global__ void cvt_transpose_W_kernel(const float* __restrict__ W,
                                       _Float16* __restrict__ Wt, int Fin) {
  int idx = blockIdx.x * 256 + threadIdx.x;
  int total = GAT_H * Fin * GAT_D;
  if (idx >= total) return;
  int d = idx & (GAT_D - 1);
  int f = (idx >> 6) % Fin;
  int h = idx / (GAT_D * Fin);
  Wt[((size_t)(h * GAT_D + d)) * Fin + f] =
      (_Float16)W[((size_t)h * Fin + f) * GAT_D + d];
}

// ---------------------------------------------------------------------------
// Projection: Wh[b,h] = X[b] (NxFin) @ W[h] (FinxD), output f16 [B][H][D][N].
// Workgroup = 128 threads = 4 waves; wave w computes the 16x16 tile
// (rows = 16 graph nodes of this block, cols = D-tile w). K-loop step 32.
// ---------------------------------------------------------------------------
__global__ __launch_bounds__(128) void proj_kernel(
    const _Float16* __restrict__ X,    // [B][N][Fin] row-major f16
    const _Float16* __restrict__ Wt,   // [H][D][Fin] f16
    _Float16* __restrict__ Wh,         // [B][H][D][N] f16
    int Fin) {
  const int ntile = blockIdx.x;            // N/16 tiles
  const int hh    = blockIdx.y;            // head
  const int b     = blockIdx.z;            // batch
  const int lane  = threadIdx.x & 31;
  const int dtile = threadIdx.x >> 5;      // 0..3 (D=64)
  const int ml    = lane & 15;             // A-row within tile / B-col within tile
  const int hi    = lane >> 4;             // half-wave select

  const _Float16* xrow =
      X + ((size_t)b * GAT_N + ntile * 16 + ml) * Fin;
  const int d = dtile * 16 + ml;
  const _Float16* wrow =
      Wt + ((size_t)(hh * GAT_D + d)) * Fin;

  v8f c = {};
  for (int kb = 0; kb < Fin; kb += 32) {
    // A fragment (contiguous k order, same as B):
    //   lane<16 holds K kb+0..15; lane>=16 holds K kb+16..31
    v8h alo = *(const v8h*)(xrow + kb + 16 * hi);
    v8h ahi = *(const v8h*)(xrow + kb + 16 * hi + 8);
    v16h a;
#pragma unroll
    for (int e = 0; e < 8; ++e) { a[e] = alo[e]; a[e + 8] = ahi[e]; }
    // B fragment (32x16): lanes 0-15 hold K kb..kb+15, lanes 16-31 hold kb+16..kb+31
    v16h bf = *(const v16h*)(wrow + kb + 16 * hi);
    c = __builtin_amdgcn_wmma_f32_16x16x32_f16(false, a, false, bf,
                                               (short)0, c, false, false);
  }
  // C layout: element r -> node row (r + 8*hi), col = ml. Store [d][n]:
  // 8 consecutive n per lane -> one 16-byte store.
  _Float16 tmp[8];
#pragma unroll
  for (int r = 0; r < 8; ++r) tmp[r] = (_Float16)c[r];
  _Float16* dst = Wh + (((size_t)b * GAT_H + hh) * GAT_D + d) * GAT_N +
                  ntile * 16 + 8 * hi;
  *(v8h*)dst = *(const v8h*)tmp;
}

// ---------------------------------------------------------------------------
// s_i = Wh . a_l, s_j = Wh . a_r  per node (length-64 dots).
// ---------------------------------------------------------------------------
__global__ void s_kernel(const _Float16* __restrict__ Wh,  // [B][H][D][N]
                         const float* __restrict__ a,      // [H][2D]
                         float* __restrict__ sI, float* __restrict__ sJ) {
  int idx = blockIdx.x * 256 + threadIdx.x;        // B*H*N
  if (idx >= GAT_B * GAT_H * GAT_N) return;
  int n  = idx & (GAT_N - 1);
  int bh = idx >> 10;
  int h  = bh % GAT_H;
  const _Float16* w = Wh + (size_t)bh * GAT_D * GAT_N + n;
  const float* al = a + h * (2 * GAT_D);
  const float* ar = al + GAT_D;
  float si = 0.f, sj = 0.f;
#pragma unroll 4
  for (int d = 0; d < GAT_D; ++d) {
    float v = (float)w[(size_t)d * GAT_N];
    si += al[d] * v;
    sj += ar[d] * v;
  }
  sI[idx] = si;
  sJ[idx] = sj;
}

// ---------------------------------------------------------------------------
// Attention + aggregation, flash-style two-pass per 16-row tile.
// Pass 1: row max + exp-sum; stores probability tile P (16x1024 f16) in LDS.
// Pass 2: out_tile = P @ Wh via WMMA (32 K-tiles), scale by 1/l, ELU/store.
// ---------------------------------------------------------------------------
__global__ __launch_bounds__(128) void agg_kernel(
    const float* __restrict__ adj,     // [N][N]
    const float* __restrict__ sI,      // [B*H][N]
    const float* __restrict__ sJ,      // [B*H][N]
    const _Float16* __restrict__ Wh,   // [B][H][D][N] f16
    _Float16* __restrict__ outF16,     // layer1: x2 f16 [B][N][HID] (or null)
    float* __restrict__ outF32,        // layer2: d_out f32 [B][N][HID] (or null)
    int doElu) {
  const int itile = blockIdx.x;
  const int hh    = blockIdx.y;
  const int b     = blockIdx.z;
  const int t     = threadIdx.x;

  __shared__ _Float16 psh[16][GAT_N + 8];  // padded stride: bank-conflict-free
  __shared__ float    sjsh[GAT_N];
  __shared__ float    red[16][8];
  __shared__ float    mrow[16];
  __shared__ float    linv[16];
  __shared__ float    srow[16];

  const size_t bh = (size_t)b * GAT_H + hh;
  const float* si = sI + bh * GAT_N;
  const float* sj = sJ + bh * GAT_N;

  if (t < 16) srow[t] = si[itile * 16 + t];
  for (int j = t; j < GAT_N; j += 128) sjsh[j] = sj[j];
  __syncthreads();

  const int irow  = t >> 3;      // 0..15 : row within tile
  const int slice = t & 7;       // 0..7  : j-stripe
  const int i     = itile * 16 + irow;
  const float s_ival = srow[irow];
  const float* adjrow = adj + (size_t)i * GAT_N;

  // ---- pass 1a: masked row max -------------------------------------------
  float mloc = -3.0e38f;
  for (int j = slice; j < GAT_N; j += 8) {
    float e = s_ival + sjsh[j];
    e = e > 0.f ? e : ALPHA * e;           // LeakyReLU
    e = adjrow[j] > 0.f ? e : NEGV;        // adjacency mask
    mloc = fmaxf(mloc, e);
  }
  red[irow][slice] = mloc;
  __syncthreads();
  if (slice == 0) {
    float mm = red[irow][0];
#pragma unroll
    for (int q = 1; q < 8; ++q) mm = fmaxf(mm, red[irow][q]);
    mrow[irow] = mm;
  }
  __syncthreads();
  const float mi = mrow[irow];

  // ---- pass 1b: exp-sum; stash P tile in LDS -----------------------------
  float lloc = 0.f;
  for (int j = slice; j < GAT_N; j += 8) {
    float e = s_ival + sjsh[j];
    e = e > 0.f ? e : ALPHA * e;
    e = adjrow[j] > 0.f ? e : NEGV;
    float p = __expf(e - mi);              // masked -> exp(-9e15 - mi) == 0
    psh[irow][j] = (_Float16)p;
    lloc += p;
  }
  red[irow][slice] = lloc;
  __syncthreads();
  if (slice == 0) {
    float ll = red[irow][0];
#pragma unroll
    for (int q = 1; q < 8; ++q) ll += red[irow][q];
    linv[irow] = 1.0f / ll;
  }
  __syncthreads();

  // ---- pass 2: out_tile = P @ Wh via WMMA --------------------------------
  const int lane  = t & 31;
  const int dtile = t >> 5;                // 0..3
  const int ml    = lane & 15;
  const int hi    = lane >> 4;
  const _Float16* whcol =
      Wh + (bh * GAT_D + dtile * 16 + ml) * GAT_N;

  v8f c = {};
  for (int jb = 0; jb < GAT_N; jb += 32) {
    // A fragment from LDS: two 16-byte ds loads per lane.
    v8h plo = *(const v8h*)(&psh[ml][jb + 16 * hi]);
    v8h phi = *(const v8h*)(&psh[ml][jb + 16 * hi + 8]);
    v16h a;
#pragma unroll
    for (int e = 0; e < 8; ++e) { a[e] = plo[e]; a[e + 8] = phi[e]; }
    // B fragment: Wh rows jb..jb+31 of this D-column, contiguous v16h load.
    v16h bf = *(const v16h*)(whcol + jb + 16 * hi);
    c = __builtin_amdgcn_wmma_f32_16x16x32_f16(false, a, false, bf,
                                               (short)0, c, false, false);
  }

  const int dglob = hh * GAT_D + dtile * 16 + ml;   // column in [0,192)
#pragma unroll
  for (int r = 0; r < 8; ++r) {
    int nrow = itile * 16 + r + 8 * hi;
    float v = c[r] * linv[r + 8 * hi];
    if (doElu) v = v > 0.f ? v : (__expf(v) - 1.0f);  // ELU between layers
    size_t off = ((size_t)b * GAT_N + nrow) * GAT_HID + dglob;
    if (outF16) outF16[off] = (_Float16)v;
    else        outF32[off] = v;
  }
}

// ---------------------------------------------------------------------------
// Workspace layout (bytes, 256-aligned)
// ---------------------------------------------------------------------------
#define OFF_HF16  ((size_t)0)          // 8*1024*64*2    = 1,048,576
#define OFF_W1T   ((size_t)1048576)    // 3*64*64*2      =    24,576
#define OFF_W2T   ((size_t)1073152)    // 3*192*64*2     =    73,728
#define OFF_WH    ((size_t)1146880)    // 24*64*1024*2   = 3,145,728 (reused L1/L2)
#define OFF_SI    ((size_t)4292608)    // 24*1024*4      =    98,304
#define OFF_SJ    ((size_t)4390912)    // 24*1024*4      =    98,304
#define OFF_X2    ((size_t)4489216)    // 8*1024*192*2   = 3,145,728
#define WS_NEEDED ((size_t)7634944)

extern "C" void kernel_launch(void* const* d_in, const int* in_sizes, int n_in,
                              void* d_out, int out_size, void* d_ws, size_t ws_size,
                              hipStream_t stream) {
  const float* h_in = (const float*)d_in[0];   // (B,N,OBS) f32
  const float* adj  = (const float*)d_in[1];   // (N,N)     f32
  const float* W1   = (const float*)d_in[2];   // (H,OBS,D) f32
  const float* a1   = (const float*)d_in[3];   // (H,2D)    f32
  const float* W2   = (const float*)d_in[4];   // (H,HID,D) f32
  const float* a2   = (const float*)d_in[5];   // (H,2D)    f32
  float* out        = (float*)d_out;           // (B,N,HID) f32

  if (ws_size < WS_NEEDED) return;
  char* ws = (char*)d_ws;
  _Float16* hf16 = (_Float16*)(ws + OFF_HF16);
  _Float16* W1t  = (_Float16*)(ws + OFF_W1T);
  _Float16* W2t  = (_Float16*)(ws + OFF_W2T);
  _Float16* Wh   = (_Float16*)(ws + OFF_WH);
  float*    sIb  = (float*)(ws + OFF_SI);
  float*    sJb  = (float*)(ws + OFF_SJ);
  _Float16* x2   = (_Float16*)(ws + OFF_X2);

  // --- prep: f16 conversions ---
  {
    int n = GAT_B * GAT_N * 64;  // OBS = 64
    cvt_f32_to_f16_kernel<<<(n + 255) / 256, 256, 0, stream>>>(h_in, hf16, n);
  }
  {
    int n = GAT_H * 64 * GAT_D;
    cvt_transpose_W_kernel<<<(n + 255) / 256, 256, 0, stream>>>(W1, W1t, 64);
  }
  {
    int n = GAT_H * GAT_HID * GAT_D;
    cvt_transpose_W_kernel<<<(n + 255) / 256, 256, 0, stream>>>(W2, W2t, GAT_HID);
  }

  dim3 gridTile(GAT_N / 16, GAT_H, GAT_B);
  int sThreads = GAT_B * GAT_H * GAT_N;

  // --- layer 1 ---
  proj_kernel<<<gridTile, 128, 0, stream>>>(hf16, W1t, Wh, 64);
  s_kernel<<<(sThreads + 255) / 256, 256, 0, stream>>>(Wh, a1, sIb, sJb);
  agg_kernel<<<gridTile, 128, 0, stream>>>(adj, sIb, sJb, Wh,
                                           x2, (float*)nullptr, /*doElu=*/1);

  // --- layer 2 (Wh buffer reused; stream order guarantees agg1 finished) ---
  proj_kernel<<<gridTile, 128, 0, stream>>>(x2, W2t, Wh, GAT_HID);
  s_kernel<<<(sThreads + 255) / 256, 256, 0, stream>>>(Wh, a2, sIb, sJb);
  agg_kernel<<<gridTile, 128, 0, stream>>>(adj, sIb, sJb, Wh,
                                           (_Float16*)nullptr, out, /*doElu=*/0);
}
